// MessageBlock_23184233464610
// MI455X (gfx1250) — matrix operand order repaired and checked by reading, not compiled, not run on hardware
//
#include <hip/hip_runtime.h>
#include <hip/hip_bf16.h>

#define F_DIM   128
#define RBF_DIM 20
#define RBF_PAD 32
#define WAVES   8
#define TILE_M  16

typedef __bf16 bf16;
typedef bf16  v16bf __attribute__((ext_vector_type(16)));
typedef bf16  v8bf  __attribute__((ext_vector_type(8)));
typedef float v8f   __attribute__((ext_vector_type(8)));

// ---------------------------------------------------------------------------
// B fragment (32x16 bf16, B = W^T): lane holds column n = lane&15, elems 0..7
// -> K = k0 + 8*(lane>>4) + j, elems 8..15 -> K = k0 + 16 + 8*(lane>>4) + j.
// Column n of B is row n of the weight matrix: two 16-byte loads per lane.
// ---------------------------------------------------------------------------
__device__ __forceinline__ v16bf load_b_frag(const bf16* __restrict__ W, int Kdim,
                                             int ncol_base, int k0, int lane) {
    const int n  = ncol_base + (lane & 15);
    const int kb = k0 + ((lane >> 4) << 3);
    v8bf lo = *reinterpret_cast<const v8bf*>(W + (size_t)n * Kdim + kb);
    v8bf hi = *reinterpret_cast<const v8bf*>(W + (size_t)n * Kdim + kb + 16);
    return __builtin_shufflevector(lo, hi, 0, 1, 2, 3, 4, 5, 6, 7,
                                   8, 9, 10, 11, 12, 13, 14, 15);
}

__device__ __forceinline__ float fast_silu(float x) {
    // x * 1/(1+e^-x) with hardware v_rcp_f32 (avoids IEEE divide expansion)
    return x * __builtin_amdgcn_rcpf(1.0f + __expf(-x));
}

// ---------------------------------------------------------------------------
// Zero-fill d_out (atomics accumulate; must re-zero every launch/replay)
// ---------------------------------------------------------------------------
__global__ void zero_out_kernel(float* __restrict__ out, int n) {
    int i = blockIdx.x * blockDim.x + threadIdx.x;
    int stride = gridDim.x * blockDim.x;
    for (; i < n; i += stride) out[i] = 0.0f;
}

// ---------------------------------------------------------------------------
// Convert weights to bf16 (and zero-pad Wrbf K: 20 -> 32) into d_ws
// ---------------------------------------------------------------------------
__global__ void prep_weights_kernel(const float* __restrict__ Ws1,
                                    const float* __restrict__ Ws2,
                                    const float* __restrict__ Wrbf,
                                    bf16* __restrict__ Ws1b,
                                    bf16* __restrict__ Ws2b,
                                    bf16* __restrict__ Wrbfb) {
    int i = blockIdx.x * blockDim.x + threadIdx.x;
    if (i < F_DIM * F_DIM)      Ws1b[i] = (bf16)Ws1[i];
    if (i < 3 * F_DIM * F_DIM)  Ws2b[i] = (bf16)Ws2[i];
    if (i < 3 * F_DIM * RBF_PAD) {
        int r = i / RBF_PAD, k = i % RBF_PAD;
        Wrbfb[i] = (bf16)((k < RBF_DIM) ? Wrbf[r * RBF_DIM + k] : 0.0f);
    }
}

// ---------------------------------------------------------------------------
// Main edge kernel: one wave32 per 16-edge tile.
// ---------------------------------------------------------------------------
__global__ void __launch_bounds__(WAVES * 32)
edge_message_kernel(const float* __restrict__ s,          // [A,128]
                    const float* __restrict__ vec,        // [A,3,128]
                    const float* __restrict__ evec,       // [E,3]
                    const float* __restrict__ edist,      // [E]
                    const float* __restrict__ erbf,       // [E,20]
                    const float* __restrict__ bs1,        // [128]
                    const float* __restrict__ bs2,        // [384]
                    const float* __restrict__ brbf,       // [384]
                    const int*   __restrict__ eidx,       // [2,E]
                    const int*   __restrict__ cutoff,     // scalar
                    const bf16*  __restrict__ Ws1b,       // [128,128]
                    const bf16*  __restrict__ Ws2b,       // [384,128]
                    const bf16*  __restrict__ Wrbfb,      // [384,32]
                    float* __restrict__ out_ds,           // [A,128]
                    float* __restrict__ out_dvec,         // [A,3,128]
                    int E) {
    __shared__ __attribute__((aligned(16))) bf16 hstage[WAVES][TILE_M * F_DIM];
    __shared__ float sc_fcut[WAVES][TILE_M];
    __shared__ float sc_vn[WAVES][3][TILE_M];
    __shared__ int   sc_dst[WAVES][TILE_M];
    __shared__ int   sc_src[WAVES][TILE_M];

    const int wave = threadIdx.x >> 5;
    const int lane = threadIdx.x & 31;
    const int tile = blockIdx.x * WAVES + wave;
    if (tile * TILE_M >= E) return;           // wave-uniform: EXEC stays full
    const int e0   = tile * TILE_M;
    const int half = lane >> 4;
    const int l15  = lane & 15;

    // ---- per-edge scalars (lanes 0..15, one edge each), staged in LDS ----
    if (lane < TILE_M) {
        int  e     = e0 + lane;
        bool valid = (e < E);
        int  ec    = valid ? e : (E - 1);
        float d    = edist[ec];
        float rc   = (float)(*cutoff);
        float fc   = 0.5f * (__cosf(3.14159265358979f * d / rc) + 1.0f);
        sc_fcut[wave][lane] = (valid && d < rc) ? fc : 0.0f;
        float inv = __builtin_amdgcn_rcpf(d);
        sc_vn[wave][0][lane] = evec[ec * 3 + 0] * inv;
        sc_vn[wave][1][lane] = evec[ec * 3 + 1] * inv;
        sc_vn[wave][2][lane] = evec[ec * 3 + 2] * inv;
        sc_dst[wave][lane] = eidx[ec];
        sc_src[wave][lane] = eidx[E + ec];
    }
    // same-wave LDS ops are in order; compiler inserts s_wait_dscnt for uses

    // ---- A1 fragments: gathered s rows, fp32 -> bf16 (16x32 A layout) ----
    // lane holds row m = lane&15; elems 0..7 -> K = k0+8*half+j,
    //                             elems 8..15 -> K = k0+16+8*half+j
    const int srcm = sc_src[wave][l15];
    const float* __restrict__ srow = s + (size_t)srcm * F_DIM;
    v16bf A1[4];
#pragma unroll
    for (int kc = 0; kc < 4; ++kc) {
        const int kb = kc * 32 + half * 8;
        v16bf a;
#pragma unroll
        for (int j = 0; j < 8; ++j) {
            a[j]     = (bf16)srow[kb + j];
            a[8 + j] = (bf16)srow[kb + 16 + j];
        }
        A1[kc] = a;
    }

    // ---- GEMM1: h = SiLU(S @ Ws1^T + bs1), stage bf16 h tile into LDS ----
    // Hoist all 4 K-chunk B fragments before the WMMA chain so the loads
    // clause together and one wait covers the chain.
#pragma unroll
    for (int nt = 0; nt < 8; ++nt) {
        v16bf bfr[4];
#pragma unroll
        for (int kc = 0; kc < 4; ++kc)
            bfr[kc] = load_b_frag(Ws1b, F_DIM, nt * 16, kc * 32, lane);
        v8f acc = {0.f, 0.f, 0.f, 0.f, 0.f, 0.f, 0.f, 0.f};
#pragma unroll
        for (int kc = 0; kc < 4; ++kc)
            acc = __builtin_amdgcn_wmma_f32_16x16x32_bf16(
                false, A1[kc], false, bfr[kc], (short)0, acc, false, false);
        const int c  = nt * 16 + l15;
        const float bias = bs1[c];
#pragma unroll
        for (int r = 0; r < 8; ++r) {
            const int mrow = r + half * 8;            // C layout row
            hstage[wave][mrow * F_DIM + c] = (bf16)fast_silu(acc[r] + bias);
        }
    }

    // ---- A2 fragments: reload h from LDS in A layout (16B vector loads) ----
    v16bf A2[4];
#pragma unroll
    for (int kc = 0; kc < 4; ++kc) {
        const int kb = kc * 32 + half * 8;
        v8bf lo = *reinterpret_cast<const v8bf*>(&hstage[wave][l15 * F_DIM + kb]);
        v8bf hi = *reinterpret_cast<const v8bf*>(&hstage[wave][l15 * F_DIM + kb + 16]);
        A2[kc] = __builtin_shufflevector(lo, hi, 0, 1, 2, 3, 4, 5, 6, 7,
                                                 8, 9, 10, 11, 12, 13, 14, 15);
    }

    // ---- RBF A fragment: 16x32 with K padded 20 -> 32 by zeros ----
    v16bf Ar;
    {
        int er = e0 + l15; if (er >= E) er = E - 1;
        const float* __restrict__ rrow = erbf + (size_t)er * RBF_DIM;
#pragma unroll
        for (int j = 0; j < 8; ++j) {
            const int k1 = half * 8 + j;
            const int k2 = 16 + half * 8 + j;
            Ar[j]     = (bf16)((k1 < RBF_DIM) ? rrow[k1] : 0.0f);
            Ar[8 + j] = (bf16)((k2 < RBF_DIM) ? rrow[k2] : 0.0f);
        }
    }

    // ---- hoist per-row epilogue scalars out of the f-tile loop ----
    float fcutR[8]; int dstR[8]; int srcR[8]; float vnR[3][8];
#pragma unroll
    for (int r = 0; r < 8; ++r) {
        const int mrow = r + half * 8;
        fcutR[r] = sc_fcut[wave][mrow];
        dstR[r]  = sc_dst[wave][mrow];
        srcR[r]  = sc_src[wave][mrow];
        vnR[0][r] = sc_vn[wave][0][mrow];
        vnR[1][r] = sc_vn[wave][1][mrow];
        vnR[2][r] = sc_vn[wave][2][mrow];
    }

    // ---- GEMM2 + RBF GEMM per 16-wide feature tile; fuse cutoff, W-mult,
    //      and scatter-add so Ws/Wvv/Wvs of the same columns are co-resident.
    for (int j = 0; j < 8; ++j) {
        v8f accP[3], accR[3];
#pragma unroll
        for (int t = 0; t < 3; ++t) {
            v16bf bfr[4];
#pragma unroll
            for (int kc = 0; kc < 4; ++kc)
                bfr[kc] = load_b_frag(Ws2b, F_DIM, t * F_DIM + j * 16, kc * 32, lane);
            v16bf br = load_b_frag(Wrbfb, RBF_PAD, t * F_DIM + j * 16, 0, lane);

            v8f a = {0.f, 0.f, 0.f, 0.f, 0.f, 0.f, 0.f, 0.f};
#pragma unroll
            for (int kc = 0; kc < 4; ++kc)
                a = __builtin_amdgcn_wmma_f32_16x16x32_bf16(
                    false, A2[kc], false, bfr[kc], (short)0, a, false, false);
            accP[t] = a;

            v8f r0 = {0.f, 0.f, 0.f, 0.f, 0.f, 0.f, 0.f, 0.f};
            accR[t] = __builtin_amdgcn_wmma_f32_16x16x32_bf16(
                false, Ar, false, br, (short)0, r0, false, false);
        }

        const int fcol = j * 16 + l15;               // feature index 0..127
        float b2[3], brf[3];
#pragma unroll
        for (int t = 0; t < 3; ++t) {
            b2[t]  = bs2[t * F_DIM + fcol];
            brf[t] = brbf[t * F_DIM + fcol];
        }

#pragma unroll
        for (int r = 0; r < 8; ++r) {
            const float fc = fcutR[r];
            const int  dst = dstR[r];
            const int  srw = srcR[r];

            const float w0 = (accR[0][r] + brf[0]) * fc;
            const float w1 = (accR[1][r] + brf[1]) * fc;
            const float w2 = (accR[2][r] + brf[2]) * fc;
            const float pWs  = (accP[0][r] + b2[0]) * w0;
            const float pWvv = (accP[1][r] + b2[1]) * w1;
            const float pWvs = (accP[2][r] + b2[2]) * w2;

            atomicAdd(&out_ds[(size_t)dst * F_DIM + fcol], pWs);

            const float* __restrict__ vrow = vec + (size_t)srw * 3 * F_DIM;
            float* __restrict__ orow = out_dvec + (size_t)dst * 3 * F_DIM;
#pragma unroll
            for (int d = 0; d < 3; ++d) {
                float dv = pWvv * vrow[d * F_DIM + fcol]
                         + vnR[d][r] * pWvs;
                atomicAdd(&orow[d * F_DIM + fcol], dv);
            }
        }
    }
}

// ---------------------------------------------------------------------------
extern "C" void kernel_launch(void* const* d_in, const int* in_sizes, int n_in,
                              void* d_out, int out_size, void* d_ws, size_t ws_size,
                              hipStream_t stream) {
    const float* s      = (const float*)d_in[0];
    const float* vec    = (const float*)d_in[1];
    const float* evec   = (const float*)d_in[2];
    const float* edist  = (const float*)d_in[3];
    const float* erbf   = (const float*)d_in[4];
    const float* Ws1    = (const float*)d_in[5];
    const float* bs1    = (const float*)d_in[6];
    const float* Ws2    = (const float*)d_in[7];
    const float* bs2    = (const float*)d_in[8];
    const float* Wrbf   = (const float*)d_in[9];
    const float* brbf   = (const float*)d_in[10];
    const int*   eidx   = (const int*)d_in[11];
    const int*   cutoff = (const int*)d_in[12];

    const int E      = in_sizes[3];                 // N_EDGES
    const int nAtoms = in_sizes[0] / F_DIM;         // N_NODES

    // scratch layout: bf16 weights
    bf16* Ws1b  = (bf16*)d_ws;                      // 128*128
    bf16* Ws2b  = Ws1b + F_DIM * F_DIM;             // 384*128
    bf16* Wrbfb = Ws2b + 3 * F_DIM * F_DIM;         // 384*32

    float* out_ds   = (float*)d_out;
    float* out_dvec = out_ds + (size_t)nAtoms * F_DIM;

    // 1) zero outputs (atomics accumulate)
    {
        int threads = 256;
        int blocks  = (out_size + threads * 8 - 1) / (threads * 8);
        if (blocks < 1) blocks = 1;
        if (blocks > 4096) blocks = 4096;
        zero_out_kernel<<<blocks, threads, 0, stream>>>(out_ds, out_size);
    }
    // 2) convert weights to bf16 into scratch
    {
        int total = 3 * F_DIM * F_DIM;              // covers all three
        prep_weights_kernel<<<(total + 255) / 256, 256, 0, stream>>>(
            Ws1, Ws2, Wrbf, Ws1b, Ws2b, Wrbfb);
    }
    // 3) main edge kernel: one wave32 per 16-edge tile
    {
        int tiles  = (E + TILE_M - 1) / TILE_M;
        int blocks = (tiles + WAVES - 1) / WAVES;
        edge_message_kernel<<<blocks, WAVES * 32, 0, stream>>>(
            s, vec, evec, edist, erbf, bs1, bs2, brbf, eidx, cutoff,
            Ws1b, Ws2b, Wrbfb, out_ds, out_dvec, E);
    }
}
